// BlindSpotBlock_7112465842440
// MI455X (gfx1250) — matrix operand order repaired and checked by reading, not compiled, not run on hardware
//
#include <hip/hip_runtime.h>

#define DEV __device__ __forceinline__

typedef __attribute__((ext_vector_type(16))) _Float16 v16h;
typedef __attribute__((ext_vector_type(8)))  _Float16 v8h;
typedef __attribute__((ext_vector_type(8)))  float    v8f;

namespace {

constexpr int Bn = 8, Hh = 256, Ww = 256, Cc = 96;
constexpr int WS = 8, SS = 4, NH = 6, CH = 16, NP = 64;
constexpr int NW = (Hh / WS) * (Ww / WS);   // 1024 windows per image
constexpr float SCALE  = 0.25f;             // CH^-0.5
constexpr float NEGBIG = -1e9f;

DEV v8f wmma16(v16h a, v16h b, v8f c) {
  // D = A(16x32 f16) x B(32x16 f16) + C(16x16 f32)
  return __builtin_amdgcn_wmma_f32_16x16x32_f16(false, a, false, b,
                                                (short)0, c, false, false);
}

// ---- vectorized fragment loaders (2 x 16B contiguous loads per lane) ----

// A fragment, full K=32, from row-major [row][K] (ld halves per row).
// ISA A layout: element e -> K = (e>>3)*16 + hf*8 + (e&7)  => two contiguous
// 8-half chunks at row*ld + hf*8 and row*ld + 16 + hf*8.
DEV v16h a_frag_rm(const _Float16* src, int ld) {
  const int l = threadIdx.x & 31;
  const int row = l & 15, hf = l >> 4;
  const _Float16* p = src + row * ld + hf * 8;
  v8h lo = *(const v8h*)(p);
  v8h hi = *(const v8h*)(p + 16);
  v16h a;
#pragma unroll
  for (int e = 0; e < 8; ++e) { a[e] = lo[e]; a[e + 8] = hi[e]; }
  return a;
}

// A fragment with K=16 valid (zero-padded to 32), from row-major [row][16].
DEV v16h a_frag_k16(const _Float16* src) {
  const int l = threadIdx.x & 31;
  const int row = l & 15, hf = l >> 4;
  v8h lo = *(const v8h*)(src + row * 16);
  v8h hi = *(const v8h*)(src + row * 16 + 8);
  v16h a;
#pragma unroll
  for (int e = 0; e < 8; ++e) { a[e] = hf ? (_Float16)0.0f : lo[e]; a[e + 8] = hf ? (_Float16)0.0f : hi[e]; }
  return a;
}

// B fragment, full K=32, from [N][K]-contiguous layout (ld halves per N).
// ISA B layout: element e -> K = (e>>3)*16 + hf*8 + (e&7)  => two 8-half chunks.
DEV v16h b_frag_rm(const _Float16* src, int ld) {
  const int l = threadIdx.x & 31;
  const int n = l & 15, hf = l >> 4;
  const _Float16* p = src + n * ld + hf * 8;
  v8h lo = *(const v8h*)(p);
  v8h hi = *(const v8h*)(p + 16);
  v16h b;
#pragma unroll
  for (int e = 0; e < 8; ++e) { b[e] = lo[e]; b[e + 8] = hi[e]; }
  return b;
}

// B fragment from a fully materialized per-lane layout [32 lanes][16 halves]
// (zeros already stored for lanes 16..31 -> no register selects needed).
DEV v16h b_frag_pad(const _Float16* src) {
  const int l = threadIdx.x & 31;
  const _Float16* p = src + l * 16;
  v8h lo = *(const v8h*)(p);
  v8h hi = *(const v8h*)(p + 8);
  v16h b;
#pragma unroll
  for (int e = 0; e < 8; ++e) { b[e] = lo[e]; b[e + 8] = hi[e]; }
  return b;
}

DEV int region1d(int p) { return (p < Hh - WS) ? 0 : ((p < Hh - SS) ? 1 : 2); }

DEV float rmax16(float v) {
#pragma unroll
  for (int m = 1; m < 16; m <<= 1) v = fmaxf(v, __shfl_xor(v, m, 32));
  return v;
}
DEV float rsum16(float v) {
#pragma unroll
  for (int m = 1; m < 16; m <<= 1) v += __shfl_xor(v, m, 32);
  return v;
}

// ===================== Kernel A: window attention + LN + proj =====================
__global__ void __launch_bounds__(256)
attn_proj_kernel(const float* __restrict__ Q, const float* __restrict__ K,
                 const float* __restrict__ V, const float* __restrict__ RPB,
                 const float* __restrict__ G, const float* __restrict__ Bb,
                 const float* __restrict__ PW, const float* __restrict__ PB,
                 float* __restrict__ inter)
{
  extern __shared__ char smem[];
  float*    qf  = (float*)smem;                    // [64][96] f32          24576
  _Float16* qh  = (_Float16*)(smem + 24576);       // [6][64][16] f16       12288
  _Float16* kB  = (_Float16*)(smem + 36864);       // [6][4][32][16] f16    24576
  _Float16* vB  = (_Float16*)(smem + 61440);       // [6][2][16][32] f16    12288
  _Float16* probs = (_Float16*)(smem + 73728);     // 8 waves * [16][64]    16384
  float*    aout  = (float*)(smem + 90112);        // [64][96] f32          24576
  _Float16* nrm   = (_Float16*)(smem + 114688);    // [64][96] f16          12288
  _Float16* wlds  = (_Float16*)(smem + 126976);    // [96][96] f16          18432
  float*    rpb   = (float*)(smem + 145408);       // [225][6] f32           5408
  // total 150816 bytes

  const int tid  = threadIdx.x;
  const int wave = tid >> 5;
  const int lane = tid & 31;
  const int n16  = lane & 15;
  const int hf   = lane >> 4;

  const int blk  = blockIdx.x;
  const int bi   = blk >> 10;        // image
  const int wIdx = blk & 1023;       // window in image
  const int wy0  = (wIdx >> 5) << 3;
  const int wx0  = (wIdx & 31) << 3;

  // ---- zero-fill the lanes 16..31 half of kB (structural K-padding zeros)
  {
    unsigned* kz = (unsigned*)kB;    // kB as [24 tiles][256 u32]; zero upper 128
    for (int i = tid; i < 24 * 128; i += 256)
      kz[(i >> 7) * 256 + 128 + (i & 127)] = 0u;
  }

  // ---- stage q/k/v window (shift-roll gather) into swizzled LDS layouts
  for (int i = tid; i < NP * Cc; i += 256) {
    int p = i / Cc, c = i - p * Cc;
    int y = (wy0 + (p >> 3) + SS) & (Hh - 1);
    int x = (wx0 + (p & 7) + SS) & (Ww - 1);
    size_t g = ((size_t)(bi * Hh + y) * Ww + x) * Cc + c;
    int h = c >> 4, cl = c & 15;
    float qv = Q[g] * SCALE;
    qf[i] = qv;
    qh[(h * 64 + p) * 16 + cl] = (_Float16)qv;          // head-major, K-contig
    int ct = p >> 4, n = p & 15;                        // B-frag layout for k
    kB[(((h << 2) | ct) * 32 + n) * 16 + cl] = (_Float16)K[g];
    int kb = p >> 5, kk = p & 31;
    vB[(((h << 1) | kb) * 16 + cl) * 32 + kk] = (_Float16)V[g];  // [N][K]-contig
  }
  for (int i = tid; i < Cc * Cc; i += 256) wlds[i] = (_Float16)PW[i];
  for (int i = tid; i < 225 * NH; i += 256) rpb[i] = RPB[i];
  __syncthreads();

  // ---- attention: 24 jobs = (head h, 16-row tile r), 3 jobs per wave
  _Float16* pw = probs + wave * 16 * 64;
  for (int t = 0; t < 3; ++t) {
    const int job = wave * 3 + t;
    const int h = job >> 2, r = job & 3;

    // logits = (q*scale) @ k^T   (K = 16, zero-padded to 32)
    v16h aq = a_frag_k16(qh + (h * 64 + r * 16) * 16);
    v8f logit[4];
#pragma unroll
    for (int c = 0; c < 4; ++c) {
      v16h bk = b_frag_pad(kB + ((h << 2) | c) * 32 * 16);
      v8f z = {};
      logit[c] = wmma16(aq, bk, z);
    }

    // + relative-position bias (LDS), + region mask, + blind-spot diagonal.
    // i = r*16 + vr + 8*hf  =>  iy = 2r + hf (uniform), ix = vr.
    // rel*NH + h = pib6[vr] - pj6, with pib6 a per-job 8-entry table.
    const int iy   = 2 * r + hf;
    const int riy3 = region1d(wy0 + iy) * 3;
    int ri8[8], pib6[8];
#pragma unroll
    for (int vr = 0; vr < 8; ++vr) {
      ri8[vr]  = riy3 + region1d(wx0 + vr);
      pib6[vr] = (iy * 15 + vr + 112) * NH + h;
    }
    const int jx    = n16 & 7;
    const int rjx   = region1d(wx0 + jx);
    const int ibase = r * 16 + (hf << 3);
#pragma unroll
    for (int c = 0; c < 4; ++c) {
      const int j   = c * 16 + n16;
      const int jy  = 2 * c + (n16 >> 3);
      const int pj6 = (jy * 15 + jx) * NH;
      const int rj  = region1d(wy0 + jy) * 3 + rjx;
#pragma unroll
      for (int vr = 0; vr < 8; ++vr) {
        const float bias = rpb[pib6[vr] - pj6];
        const float m = ((ibase + vr) == j || ri8[vr] != rj) ? NEGBIG : 0.0f;
        logit[c][vr] += bias + m;
      }
    }

    // softmax across 64 cols (4 tiles x 16 lanes)
#pragma unroll
    for (int vr = 0; vr < 8; ++vr) {
      float mx = -3.0e38f;
#pragma unroll
      for (int c = 0; c < 4; ++c) mx = fmaxf(mx, logit[c][vr]);
      mx = rmax16(mx);
      float ex[4], s = 0.0f;
#pragma unroll
      for (int c = 0; c < 4; ++c) { ex[c] = __expf(logit[c][vr] - mx); s += ex[c]; }
      s = rsum16(s);
      const float inv = 1.0f / s;
      const int M = vr + (hf << 3);
#pragma unroll
      for (int c = 0; c < 4; ++c)
        pw[M * 64 + c * 16 + n16] = (_Float16)(ex[c] * inv);
    }

    // out = probs @ v + q  (K = 64 -> two chained K=32 WMMAs)
    v8f o = {};
    o = wmma16(a_frag_rm(pw,      64), b_frag_rm(vB + ((h << 1) | 0) * 16 * 32, 32), o);
    o = wmma16(a_frag_rm(pw + 32, 64), b_frag_rm(vB + ((h << 1) | 1) * 16 * 32, 32), o);
#pragma unroll
    for (int vr = 0; vr < 8; ++vr) {
      const int i = r * 16 + vr + (hf << 3);
      aout[i * Cc + h * CH + n16] = o[vr] + qf[i * Cc + h * CH + n16];
    }
  }
  __syncthreads();

  // ---- LayerNorm over 96 channels, 4 lanes per row
  {
    const int row = tid >> 2, part = tid & 3;
    const float* xr = aout + row * Cc + part * 24;
    float s = 0.0f, sq = 0.0f;
#pragma unroll
    for (int c = 0; c < 24; ++c) { float v = xr[c]; s += v; sq += v * v; }
    s  += __shfl_xor(s, 1, 32);  s  += __shfl_xor(s, 2, 32);
    sq += __shfl_xor(sq, 1, 32); sq += __shfl_xor(sq, 2, 32);
    const float mu   = s * (1.0f / 96.0f);
    const float var  = sq * (1.0f / 96.0f) - mu * mu;
    const float rstd = rsqrtf(var + 1e-5f);
#pragma unroll
    for (int c = 0; c < 24; ++c) {
      const int ch = part * 24 + c;
      nrm[row * Cc + ch] = (_Float16)((xr[c] - mu) * rstd * G[ch] + Bb[ch]);
    }
  }
  __syncthreads();

  // ---- proj: (64x96) @ PW^T + PB, store with window-reverse + un-roll
  for (int t = 0; t < 3; ++t) {
    const int tile = wave * 3 + t;      // 24 tiles = 4 x 6
    const int mt = tile / 6, nt = tile % 6;
    v8f acc = {};
#pragma unroll
    for (int kb = 0; kb < 3; ++kb) {
      v16h a = a_frag_rm(nrm  + (mt * 16) * Cc + kb * 32, Cc);
      v16h b = b_frag_rm(wlds + (nt * 16) * Cc + kb * 32, Cc);
      acc = wmma16(a, b, acc);
    }
    const int o  = nt * 16 + n16;
    const float pb = PB[o];
#pragma unroll
    for (int vr = 0; vr < 8; ++vr) {
      const int i = mt * 16 + vr + (hf << 3);
      const int y = (wy0 + (i >> 3) + SS) & (Hh - 1);
      const int x = (wx0 + (i & 7) + SS) & (Ww - 1);
      inter[((size_t)(bi * Hh + y) * Ww + x) * Cc + o] = acc[vr] + pb;
    }
  }
}

// ===================== Kernel B: LN2 -> fc1 -> GELU -> fc2 + residual =====================
__global__ void __launch_bounds__(256)
mlp_kernel(const float* __restrict__ inter, const float* __restrict__ G2,
           const float* __restrict__ B2, const float* __restrict__ W1,
           const float* __restrict__ b1, const float* __restrict__ W2,
           const float* __restrict__ b2, float* __restrict__ out)
{
  extern __shared__ char smem[];
  float*    x_lds = (float*)smem;                 // 64*96 f32 = 24576
  _Float16* nrm   = (_Float16*)(smem + 24576);    // 12288
  _Float16* hbuf  = (_Float16*)(smem + 36864);    // 12288
  _Float16* wlds  = (_Float16*)(smem + 49152);    // 18432 -> total 67584

  const int tid  = threadIdx.x;
  const int wave = tid >> 5;
  const int lane = tid & 31;
  const int n16  = lane & 15;
  const int hf   = lane >> 4;
  const size_t tok0 = (size_t)blockIdx.x * 64;

  for (int i = tid; i < 64 * Cc; i += 256) x_lds[i] = inter[tok0 * Cc + i];
  for (int i = tid; i < Cc * Cc; i += 256) wlds[i] = (_Float16)W1[i];
  __syncthreads();

  // LN2
  {
    const int row = tid >> 2, part = tid & 3;
    const float* xr = x_lds + row * Cc + part * 24;
    float s = 0.0f, sq = 0.0f;
#pragma unroll
    for (int c = 0; c < 24; ++c) { float v = xr[c]; s += v; sq += v * v; }
    s  += __shfl_xor(s, 1, 32);  s  += __shfl_xor(s, 2, 32);
    sq += __shfl_xor(sq, 1, 32); sq += __shfl_xor(sq, 2, 32);
    const float mu   = s * (1.0f / 96.0f);
    const float var  = sq * (1.0f / 96.0f) - mu * mu;
    const float rstd = rsqrtf(var + 1e-5f);
#pragma unroll
    for (int c = 0; c < 24; ++c) {
      const int ch = part * 24 + c;
      nrm[row * Cc + ch] = (_Float16)((xr[c] - mu) * rstd * G2[ch] + B2[ch]);
    }
  }
  __syncthreads();

  // fc1 + exact GELU
  for (int t = 0; t < 3; ++t) {
    const int tile = wave * 3 + t;
    const int mt = tile / 6, nt = tile % 6;
    v8f acc = {};
#pragma unroll
    for (int kb = 0; kb < 3; ++kb)
      acc = wmma16(a_frag_rm(nrm  + (mt * 16) * Cc + kb * 32, Cc),
                   b_frag_rm(wlds + (nt * 16) * Cc + kb * 32, Cc), acc);
    const int o = nt * 16 + n16;
    const float bb = b1[o];
#pragma unroll
    for (int vr = 0; vr < 8; ++vr) {
      const float xv = acc[vr] + bb;
      const float gl = 0.5f * xv * (1.0f + erff(xv * 0.70710678118654752f));
      const int i = mt * 16 + vr + (hf << 3);
      hbuf[i * Cc + o] = (_Float16)gl;
    }
  }
  __syncthreads();
  for (int i = tid; i < Cc * Cc; i += 256) wlds[i] = (_Float16)W2[i];
  __syncthreads();

  // fc2 + residual
  for (int t = 0; t < 3; ++t) {
    const int tile = wave * 3 + t;
    const int mt = tile / 6, nt = tile % 6;
    v8f acc = {};
#pragma unroll
    for (int kb = 0; kb < 3; ++kb)
      acc = wmma16(a_frag_rm(hbuf + (mt * 16) * Cc + kb * 32, Cc),
                   b_frag_rm(wlds + (nt * 16) * Cc + kb * 32, Cc), acc);
    const int o = nt * 16 + n16;
    const float bb = b2[o];
#pragma unroll
    for (int vr = 0; vr < 8; ++vr) {
      const int i = mt * 16 + vr + (hf << 3);
      out[(tok0 + i) * Cc + o] = x_lds[i * Cc + o] + acc[vr] + bb;
    }
  }
}

} // namespace

extern "C" void kernel_launch(void* const* d_in, const int* in_sizes, int n_in,
                              void* d_out, int out_size, void* d_ws, size_t ws_size,
                              hipStream_t stream) {
  (void)in_sizes; (void)n_in; (void)out_size; (void)ws_size;
  const float* Q   = (const float*)d_in[0];
  const float* K   = (const float*)d_in[1];
  const float* V   = (const float*)d_in[2];
  const float* RPB = (const float*)d_in[3];
  const float* G   = (const float*)d_in[4];
  const float* Bb  = (const float*)d_in[5];
  const float* PW  = (const float*)d_in[6];
  const float* PB  = (const float*)d_in[7];
  const float* G2  = (const float*)d_in[8];
  const float* B2  = (const float*)d_in[9];
  const float* W1  = (const float*)d_in[10];
  const float* b1  = (const float*)d_in[11];
  const float* W2  = (const float*)d_in[12];
  const float* b2  = (const float*)d_in[13];
  float* inter = (float*)d_ws;                 // (B,H,W,C) f32 intermediate

  attn_proj_kernel<<<Bn * NW, 256, 150816, stream>>>(Q, K, V, RPB, G, Bb, PW, PB, inter);
  mlp_kernel<<<(Bn * Hh * Ww) / 64, 256, 67584, stream>>>(inter, G2, B2, W1, b1, W2, b2,
                                                          (float*)d_out);
}
